// Model_79044578116082
// MI455X (gfx1250) — matrix operand order repaired and checked by reading, not compiled, not run on hardware
//
#include <hip/hip_runtime.h>
#include <hip/hip_bf16.h>
#include <math.h>

// ---------------------------------------------------------------------------
// CDNA5 (gfx1250) WMMA types
// ---------------------------------------------------------------------------
typedef __attribute__((ext_vector_type(16))) _Float16 v16h;
typedef __attribute__((ext_vector_type(8)))  float    v8f;
typedef __attribute__((ext_vector_type(8)))  _Float16 h8;
// float4 vector that only promises 4-byte alignment (rows may start at any
// dword when K is odd, e.g. K=19127); AMDGPU multi-dword loads only need
// dword alignment, so this still lowers to global_load_b128.
typedef float f4v __attribute__((ext_vector_type(4), aligned(4)));

union H8U { h8 v; _Float16 e[8]; };
union CFrag { v8f v; float e[8]; };

// ---------------------------------------------------------------------------
// Tiled GEMM on v_wmma_f32_16x16x32_f16.
//   C[M,N] = act( scale * (A[M,K] @ B) + bias )
//   BT=false: B is [K,N] row-major.  BT=true: B is [N,K] row-major (B^T).
//
// Block = 256 threads = 8 waves; 64x64 C tile, K step 32.
// Wave (wy, wxg) computes one 16-row tile x two 16-col tiles: one A fragment
// is reused across TWO independent back-to-back WMMAs (better XDL pipelining,
// half the A-side LDS traffic per matrix op).
// LDS tiles are stored in FRAGMENT ORDER ([tile][lane][16 halves]) so each
// lane's WMMA operand is one contiguous 32B LDS load (2x ds_load_b128).
// Double-buffered LDS overlaps next tile's global loads with current WMMA.
// All bounds handling is block-uniform (scalar branches) or clamp+select —
// no per-element exec-mask branching around loads in the hot path.
// act: 0=none, 1=relu, 2=tanh, 3=abs(tanh)
// ---------------------------------------------------------------------------
template <bool BT>
__global__ __launch_bounds__(256) void gemm_wmma(
    const float* __restrict__ A, const float* __restrict__ B,
    const float* __restrict__ bias, float* __restrict__ C,
    int M, int N, int K, int act, float scale)
{
    // Fragment-ordered staging buffers, double buffered.
    __shared__ _Float16 sAf[2][4 * 32 * 16];  // [buf][rowTile 0..3][lane][i]
    __shared__ _Float16 sBf[2][4 * 32 * 16];  // [buf][colTile 0..3][lane][i]

    const int tid  = threadIdx.x;
    const int lane = tid & 31;
    const int wave = tid >> 5;
    const int wy   = wave >> 1;          // 0..3 row tile
    const int ct0  = (wave & 1) * 2;     // first of two col tiles (0 or 2)
    const int bm   = blockIdx.y * 64;
    const int bn   = blockIdx.x * 64;

    // ---- A staging geometry: thread -> (row 0..63, k-octet 0,8,16,24) ----
    // ISA 16-bit A 16x32 layout: lanes 0-15 hold K {0..7,16..23} (i=0..15),
    // lanes 16-31 hold K {8..15,24..31}.
    const int ar     = tid >> 2;                       // 0..63
    const int ac     = (tid & 3) * 8;                  // 0,8,16,24
    const int a_lane = (ar & 15) + ((ac & 8) ? 16 : 0);
    const int a_half = (ac & 16) ? 8 : 0;
    const int a_off  = (ar >> 4) * 512 + a_lane * 16 + a_half; // in halves

    // ---- B staging geometry: thread -> (col 0..63, k-octet 0,8,16,24) ----
    // ISA 16-bit B 32x16 layout: lanes 0-15 = cols, K {0..7,16..23}; lanes 16-31 K {8..15,24..31}.
    const int bcol   = tid & 63;
    const int bkq    = (tid >> 6) * 8;                 // 0,8,16,24
    const int b_lane = (bcol & 15) + ((bkq & 8) ? 16 : 0);
    const int b_off  = (bcol >> 4) * 512 + b_lane * 16 + ((bkq & 16) ? 8 : 0);

    const bool intA = (bm + 64 <= M);   // block-uniform
    const bool intB = (bn + 64 <= N);   // block-uniform

    // ---- staging helpers (uniform fast/slow selection) ----
    auto stageA = [&](int sb, int k0) {
        H8U pa;
        if (intA && (k0 + 32 <= K)) {
            const float* ap = A + (size_t)(bm + ar) * K + (k0 + ac);
            f4v v0 = *(const f4v*)ap;
            f4v v1 = *(const f4v*)(ap + 4);
            #pragma unroll
            for (int j = 0; j < 4; ++j) {
                pa.e[j]     = (_Float16)v0[j];
                pa.e[4 + j] = (_Float16)v1[j];
            }
        } else {
            #pragma unroll
            for (int j = 0; j < 8; ++j) {
                int r = bm + ar, c = k0 + ac + j;
                int rc = r < M ? r : M - 1;
                int cc = c < K ? c : K - 1;
                float v = A[(size_t)rc * K + cc];      // always safe
                pa.e[j] = (_Float16)((r < M && c < K) ? v : 0.0f);
            }
        }
        *(h8*)&sAf[sb][a_off] = pa.v;
    };
    auto stageB = [&](int sb, int k0) {
        H8U pb;
        if (intB && (k0 + 32 <= K)) {
            if (BT) {
                const float* bp = B + (size_t)(bn + bcol) * K + (k0 + bkq);
                f4v v0 = *(const f4v*)bp;
                f4v v1 = *(const f4v*)(bp + 4);
                #pragma unroll
                for (int j = 0; j < 4; ++j) {
                    pb.e[j]     = (_Float16)v0[j];
                    pb.e[4 + j] = (_Float16)v1[j];
                }
            } else {
                #pragma unroll
                for (int j = 0; j < 8; ++j)
                    pb.e[j] = (_Float16)B[(size_t)(k0 + bkq + j) * N + (bn + bcol)];
            }
        } else {
            #pragma unroll
            for (int j = 0; j < 8; ++j) {
                int kk = k0 + bkq + j, nn = bn + bcol;
                int kc = kk < K ? kk : K - 1;
                int nc = nn < N ? nn : N - 1;
                float v = BT ? B[(size_t)nc * K + kc] : B[(size_t)kc * N + nc];
                pb.e[j] = (_Float16)((kk < K && nn < N) ? v : 0.0f);
            }
        }
        *(h8*)&sBf[sb][b_off] = pb.v;
    };

    v8f acc0 = {};
    v8f acc1 = {};
    stageA(0, 0);
    stageB(0, 0);
    __syncthreads();

    int buf = 0;
    for (int k0 = 0; k0 < K; k0 += 32) {
        if (k0 + 64 < K)   // uniform; speculative prefetch is drop-safe
            __builtin_prefetch(A + (size_t)(bm + ar) * K + (k0 + 64 + ac), 0, 0);
        if (k0 + 32 < K) { // stage next tile into the other buffer
            stageA(buf ^ 1, k0 + 32);
            stageB(buf ^ 1, k0 + 32);
        }
        // Contiguous 32B fragment loads (2x ds_load_b128 each); one A frag
        // feeds two back-to-back independent WMMAs.
        v16h af  = *(const v16h*)&sAf[buf][wy * 512 + lane * 16];
        v16h bf0 = *(const v16h*)&sBf[buf][ct0 * 512 + lane * 16];
        v16h bf1 = *(const v16h*)&sBf[buf][(ct0 + 1) * 512 + lane * 16];
        acc0 = __builtin_amdgcn_wmma_f32_16x16x32_f16(
            false, af, false, bf0, (short)0, acc0, false, false);
        acc1 = __builtin_amdgcn_wmma_f32_16x16x32_f16(
            false, af, false, bf1, (short)0, acc1, false, false);
        __syncthreads();
        buf ^= 1;
    }

    // Epilogue. C/D layout: VGPR j -> M = j + 8*(lane>=16), N = lane&15.
    CFrag cf0; cf0.v = acc0;
    CFrag cf1; cf1.v = acc1;
    const int mo = (lane >> 4) * 8;
    #pragma unroll
    for (int t = 0; t < 2; ++t) {
        const CFrag& cf = t ? cf1 : cf0;
        const int n = bn + (ct0 + t) * 16 + (lane & 15);
        if (n < N) {
            float bv = bias ? bias[n] : 0.0f;
            #pragma unroll
            for (int j = 0; j < 8; ++j) {
                int m = bm + wy * 16 + mo + j;
                if (m < M) {
                    float v = cf.e[j] * scale + bv;
                    if (act == 1)      v = fmaxf(v, 0.0f);
                    else if (act == 2) v = tanhf(v);
                    else if (act == 3) v = fabsf(tanhf(v));
                    C[(size_t)m * N + n] = v;
                }
            }
        }
    }
}

// ---------------------------------------------------------------------------
// Elementwise / scatter helpers for the GAT attention path (memory bound).
// ---------------------------------------------------------------------------
__global__ void fill_kernel(float* p, float v, size_t n) {
    size_t i = blockIdx.x * (size_t)blockDim.x + threadIdx.x;
    if (i < n) p[i] = v;
}

// edge e in [0, E+N): first E from edge_index, last N are self loops.
__device__ __forceinline__ void edge_sd(const int* ei, int E, int e, int& s, int& d) {
    if (e < E) { s = ei[e]; d = ei[E + e]; }
    else       { s = e - E; d = e - E; }
}

__device__ __forceinline__ void atomicMaxF(float* addr, float v) {
    if (v >= 0.0f) atomicMax((int*)addr, __float_as_int(v));
    else           atomicMin((unsigned int*)addr, __float_as_uint(v));
}

// al_s/al_d: per-node, per-head attention logits (sum over channels).
__global__ void attn_logits(const float* __restrict__ L,
                            const float* __restrict__ a_s,
                            const float* __restrict__ a_d,
                            float* __restrict__ al_s, float* __restrict__ al_d,
                            int N, int H, int Cc) {
    size_t i = blockIdx.x * (size_t)blockDim.x + threadIdx.x;
    if (i >= (size_t)N * H) return;
    int h = (int)(i % H);
    size_t node = i / H;
    const float* row = L + node * (size_t)H * Cc + (size_t)h * Cc;
    float ss = 0.0f, sd = 0.0f;
    for (int c = 0; c < Cc; ++c) {
        float v = row[c];
        ss += v * a_s[h * Cc + c];
        sd += v * a_d[h * Cc + c];
    }
    al_s[i] = ss;
    al_d[i] = sd;
}

__global__ void edge_max(const float* __restrict__ al_s, const float* __restrict__ al_d,
                         const int* __restrict__ ei, int E, int N, int H,
                         float* __restrict__ m) {
    size_t i = blockIdx.x * (size_t)blockDim.x + threadIdx.x;
    size_t tot = (size_t)(E + N) * H;
    if (i >= tot) return;
    int h = (int)(i % H);
    int e = (int)(i / H);
    int s, d; edge_sd(ei, E, e, s, d);
    float v = al_s[(size_t)s * H + h] + al_d[(size_t)d * H + h];
    v = (v > 0.0f) ? v : 0.2f * v;   // leaky_relu(0.2)
    atomicMaxF(&m[(size_t)d * H + h], v);
}

__global__ void edge_exp(const float* __restrict__ al_s, const float* __restrict__ al_d,
                         const int* __restrict__ ei, int E, int N, int H,
                         const float* __restrict__ m, float* __restrict__ p,
                         float* __restrict__ ssum) {
    size_t i = blockIdx.x * (size_t)blockDim.x + threadIdx.x;
    size_t tot = (size_t)(E + N) * H;
    if (i >= tot) return;
    int h = (int)(i % H);
    int e = (int)(i / H);
    int s, d; edge_sd(ei, E, e, s, d);
    float v = al_s[(size_t)s * H + h] + al_d[(size_t)d * H + h];
    v = (v > 0.0f) ? v : 0.2f * v;
    float pe = __expf(v - m[(size_t)d * H + h]);
    p[i] = pe;
    atomicAdd(&ssum[(size_t)d * H + h], pe);
}

__global__ void edge_alpha(float* __restrict__ p, const float* __restrict__ ssum,
                           const int* __restrict__ ei, int E, int N, int H) {
    size_t i = blockIdx.x * (size_t)blockDim.x + threadIdx.x;
    size_t tot = (size_t)(E + N) * H;
    if (i >= tot) return;
    int h = (int)(i % H);
    int e = (int)(i / H);
    int s, d; edge_sd(ei, E, e, s, d);
    p[i] /= ssum[(size_t)d * H + h];
}

// out[dst] += L[src] * alpha[e,h]   (one block per edge)
__global__ void gat_scatter(const float* __restrict__ L, const float* __restrict__ alpha,
                            const int* __restrict__ ei, int E, int N, int H, int Cc,
                            float* __restrict__ out) {
    int e = blockIdx.x;
    int s, d; edge_sd(ei, E, e, s, d);
    int HC = H * Cc;
    const float* src = L + (size_t)s * HC;
    float* dst = out + (size_t)d * HC;
    for (int i = threadIdx.x; i < HC; i += blockDim.x) {
        int h = i / Cc;
        atomicAdd(&dst[i], src[i] * alpha[(size_t)e * H + h]);
    }
}

__global__ void bias_relu(float* __restrict__ x, const float* __restrict__ b,
                          int N, int HC) {
    size_t i = blockIdx.x * (size_t)blockDim.x + threadIdx.x;
    if (i >= (size_t)N * HC) return;
    float v = x[i] + b[i % HC];
    x[i] = fmaxf(v, 0.0f);
}

// global_max_pool: batch = arange(N)//nodes_per_graph (contiguous groups).
__global__ void pool_max(const float* __restrict__ h, float* __restrict__ out,
                         int nb, int npg, int F) {
    size_t i = blockIdx.x * (size_t)blockDim.x + threadIdx.x;
    if (i >= (size_t)nb * F) return;
    int c = (int)(i % F);
    int g = (int)(i / F);
    const float* base = h + (size_t)g * npg * F + c;
    float m = base[0];
    for (int k = 1; k < npg; ++k) m = fmaxf(m, base[(size_t)k * F]);
    out[i] = m;
}

__global__ void copy_cols(const float* __restrict__ src, int src_ld,
                          float* __restrict__ dst, int dst_ld, int col_off,
                          int rows, int cols) {
    size_t i = blockIdx.x * (size_t)blockDim.x + threadIdx.x;
    if (i >= (size_t)rows * cols) return;
    int c = (int)(i % cols);
    int r = (int)(i / cols);
    dst[(size_t)r * dst_ld + col_off + c] = src[(size_t)r * src_ld + c];
}

__global__ void row_norm(const float* __restrict__ in, float* __restrict__ out,
                         int rows, int cols) {
    int r = blockIdx.x * blockDim.x + threadIdx.x;
    if (r >= rows) return;
    const float* p = in + (size_t)r * cols;
    float ss = 0.0f;
    for (int c = 0; c < cols; ++c) ss += p[c] * p[c];
    float nrm = fmaxf(sqrtf(ss), 1e-12f);
    float inv = 1.0f / nrm;
    float* q = out + (size_t)r * cols;
    for (int c = 0; c < cols; ++c) q[c] = p[c] * inv;
}

// ---------------------------------------------------------------------------
// Host-side orchestration helpers
// ---------------------------------------------------------------------------
static inline void launch_gemm(const float* A, const float* B, const float* bias,
                               float* C, int M, int N, int K, int act, float scale,
                               bool bt, hipStream_t stream) {
    dim3 grid((N + 63) / 64, (M + 63) / 64);
    if (bt) gemm_wmma<true ><<<grid, 256, 0, stream>>>(A, B, bias, C, M, N, K, act, scale);
    else    gemm_wmma<false><<<grid, 256, 0, stream>>>(A, B, bias, C, M, N, K, act, scale);
}

static inline size_t ceil_div(size_t a, size_t b) { return (a + b - 1) / b; }

static void gat_layer(const float* X, float* L, float* ACC,
                      const float* W, const float* a_s, const float* a_d, const float* b,
                      const int* ei, int E, int N, int Fin, int H, int Cc,
                      float* al_s, float* al_d, float* mbuf, float* sbuf, float* pbuf,
                      hipStream_t stream) {
    const int HC = H * Cc;
    // h = X @ W  (WMMA)
    launch_gemm(X, W, nullptr, L, N, HC, Fin, /*act*/0, 1.0f, false, stream);

    const size_t nh  = (size_t)N * H;
    const size_t nhc = (size_t)N * HC;
    fill_kernel<<<ceil_div(nh, 256),  256, 0, stream>>>(mbuf, -INFINITY, nh);
    fill_kernel<<<ceil_div(nh, 256),  256, 0, stream>>>(sbuf, 0.0f, nh);
    fill_kernel<<<ceil_div(nhc, 256), 256, 0, stream>>>(ACC,  0.0f, nhc);

    attn_logits<<<ceil_div(nh, 256), 256, 0, stream>>>(L, a_s, a_d, al_s, al_d, N, H, Cc);

    const int Et = E + N;
    const size_t eh = (size_t)Et * H;
    edge_max  <<<ceil_div(eh, 256), 256, 0, stream>>>(al_s, al_d, ei, E, N, H, mbuf);
    edge_exp  <<<ceil_div(eh, 256), 256, 0, stream>>>(al_s, al_d, ei, E, N, H, mbuf, pbuf, sbuf);
    edge_alpha<<<ceil_div(eh, 256), 256, 0, stream>>>(pbuf, sbuf, ei, E, N, H);
    gat_scatter<<<Et, 256, 0, stream>>>(L, pbuf, ei, E, N, H, Cc, ACC);
    bias_relu<<<ceil_div(nhc, 256), 256, 0, stream>>>(ACC, b, N, HC);
}

// ---------------------------------------------------------------------------
// Entry point
// ---------------------------------------------------------------------------
extern "C" void kernel_launch(void* const* d_in, const int* in_sizes, int n_in,
                              void* d_out, int out_size, void* d_ws, size_t ws_size,
                              hipStream_t stream) {
    // ---- inputs (setup_inputs dict order) ----
    const float* x    = (const float*)d_in[0];     // [N,109]
    const int*   ei   = (const int*)  d_in[1];     // [2,E]
    const float* w    = (const float*)d_in[3];     // [B,750]
    const float* z    = (const float*)d_in[4];     // [B,19127]
    const float* v    = (const float*)d_in[5];     // [B,1024]
    const float* side = (const float*)d_in[6];     // [994,1050]
    const float* g_W [4] = {(const float*)d_in[7],  (const float*)d_in[11],
                            (const float*)d_in[15], (const float*)d_in[19]};
    const float* g_as[4] = {(const float*)d_in[8],  (const float*)d_in[12],
                            (const float*)d_in[16], (const float*)d_in[20]};
    const float* g_ad[4] = {(const float*)d_in[9],  (const float*)d_in[13],
                            (const float*)d_in[17], (const float*)d_in[21]};
    const float* g_b [4] = {(const float*)d_in[10], (const float*)d_in[14],
                            (const float*)d_in[18], (const float*)d_in[22]};
    const float* x5_W = (const float*)d_in[23]; const float* x5_b = (const float*)d_in[24];
    const float* x6_W = (const float*)d_in[25]; const float* x6_b = (const float*)d_in[26];
    const float* w1_W = (const float*)d_in[27]; const float* w1_b = (const float*)d_in[28];
    const float* w2_W = (const float*)d_in[29]; const float* w2_b = (const float*)d_in[30];
    const float* z1_W = (const float*)d_in[31]; const float* z1_b = (const float*)d_in[32];
    const float* z2_W = (const float*)d_in[33]; const float* z2_b = (const float*)d_in[34];
    const float* v1_W = (const float*)d_in[35]; const float* v1_b = (const float*)d_in[36];
    const float* v2_W = (const float*)d_in[37]; const float* v2_b = (const float*)d_in[38];
    const float* agg_W= (const float*)d_in[39]; const float* agg_b= (const float*)d_in[40];
    const float* col_W= (const float*)d_in[41]; const float* col_b= (const float*)d_in[42];
    const float* s1_W = (const float*)d_in[43]; const float* s1_b = (const float*)d_in[44];
    const float* s2_W = (const float*)d_in[45]; const float* s2_b = (const float*)d_in[46];

    const int N  = in_sizes[0] / 109;   // 32768
    const int E  = in_sizes[1] / 2;     // 131072
    const int Bn = 1024;                // graphs
    const int NS = 994, SF = 1050;

    // ---- workspace carve ----
    char* wp = (char*)d_ws;
    auto alloc = [&](size_t bytes) -> float* {
        float* r = (float*)wp;
        wp += (bytes + 255) & ~(size_t)255;
        return r;
    };
    float* hA    = alloc((size_t)N * 1024 * 4);   // GAT linear output (L)
    float* hB    = alloc((size_t)N * 1024 * 4);   // GAT aggregate / features (X)
    float* al_s  = alloc((size_t)N * 8 * 4);
    float* al_d  = alloc((size_t)N * 8 * 4);
    float* mbuf  = alloc((size_t)N * 8 * 4);
    float* sbuf  = alloc((size_t)N * 8 * 4);
    float* pbuf  = alloc((size_t)(E + N) * 8 * 4);
    float* t1600 = alloc((size_t)Bn * 1600 * 4);
    float* t750  = alloc((size_t)Bn * 750 * 4);
    float* t256  = alloc((size_t)Bn * 256 * 4);
    float* xg    = alloc((size_t)Bn * 128 * 4);
    float* xg1   = alloc((size_t)Bn * 64 * 4);
    float* xg2   = alloc((size_t)Bn * 64 * 4);
    float* wv    = alloc((size_t)Bn * 64 * 4);
    float* zv    = alloc((size_t)Bn * 64 * 4);
    float* vv    = alloc((size_t)Bn * 64 * 4);
    float* xwz   = alloc((size_t)Bn * 64 * 4);
    float* cat192= alloc((size_t)Bn * 192 * 4);
    float* cat128= alloc((size_t)Bn * 128 * 4);
    float* drug  = alloc((size_t)Bn * 64 * 4);
    float* se1   = alloc((size_t)NS * 64 * 4);
    float* se    = alloc((size_t)NS * 64 * 4);

    // ---- 4 GAT layers (self loops handled implicitly in edge kernels) ----
    gat_layer(x,  hA, hB, g_W[0], g_as[0], g_ad[0], g_b[0], ei, E, N, 109,  8,  96,
              al_s, al_d, mbuf, sbuf, pbuf, stream);
    gat_layer(hB, hA, hB, g_W[1], g_as[1], g_ad[1], g_b[1], ei, E, N, 768,  8, 128,
              al_s, al_d, mbuf, sbuf, pbuf, stream);
    gat_layer(hB, hA, hB, g_W[2], g_as[2], g_ad[2], g_b[2], ei, E, N, 1024, 8, 128,
              al_s, al_d, mbuf, sbuf, pbuf, stream);
    gat_layer(hB, hA, hB, g_W[3], g_as[3], g_ad[3], g_b[3], ei, E, N, 1024, 1, 128,
              al_s, al_d, mbuf, sbuf, pbuf, stream);

    // ---- global max pool (32 contiguous nodes per graph) ----
    pool_max<<<ceil_div((size_t)Bn * 128, 256), 256, 0, stream>>>(hB, xg, Bn, N / Bn, 128);

    // ---- MLP towers (all WMMA GEMMs) ----
    launch_gemm(xg,   x5_W, x5_b, xg1,   Bn, 64,   128,   1, 1.0f, false, stream);
    launch_gemm(xg1,  x6_W, x6_b, xg2,   Bn, 64,   64,    1, 1.0f, false, stream);
    launch_gemm(w,    w1_W, w1_b, t750,  Bn, 750,  750,   1, 1.0f, false, stream);
    launch_gemm(t750, w2_W, w2_b, wv,    Bn, 64,   750,   1, 1.0f, false, stream);
    launch_gemm(z,    z1_W, z1_b, t1600, Bn, 1600, 19127, 1, 1.0f, false, stream);
    launch_gemm(t1600,z2_W, z2_b, zv,    Bn, 64,   1600,  1, 1.0f, false, stream);
    launch_gemm(v,    v1_W, v1_b, t256,  Bn, 256,  1024,  1, 1.0f, false, stream);
    launch_gemm(t256, v2_W, v2_b, vv,    Bn, 64,   256,   1, 1.0f, false, stream);

    // ---- concat + agg + col ----
    copy_cols<<<ceil_div((size_t)Bn * 64, 256), 256, 0, stream>>>(xg2, 64, cat192, 192, 0,   Bn, 64);
    copy_cols<<<ceil_div((size_t)Bn * 64, 256), 256, 0, stream>>>(wv,  64, cat192, 192, 64,  Bn, 64);
    copy_cols<<<ceil_div((size_t)Bn * 64, 256), 256, 0, stream>>>(zv,  64, cat192, 192, 128, Bn, 64);
    launch_gemm(cat192, agg_W, agg_b, xwz, Bn, 64, 192, 0, 1.0f, false, stream);
    copy_cols<<<ceil_div((size_t)Bn * 64, 256), 256, 0, stream>>>(xwz, 64, cat128, 128, 0,  Bn, 64);
    copy_cols<<<ceil_div((size_t)Bn * 64, 256), 256, 0, stream>>>(vv,  64, cat128, 128, 64, Bn, 64);
    launch_gemm(cat128, col_W, col_b, drug, Bn, 64, 128, /*abs(tanh)*/3, 1.0f, false, stream);

    // ---- side-effect tower ----
    launch_gemm(side, s1_W, s1_b, se1, NS, 64, SF, 1, 1.0f, false, stream);
    launch_gemm(se1,  s2_W, s2_b, se,  NS, 64, 64, /*tanh*/2, 1.0f, false, stream);

    // ---- normalize, write dn/sn into d_out, then freq = 5 * dn @ sn^T ----
    float* out  = (float*)d_out;
    float* freq = out;                              // [1024, 994]
    float* dn   = out + (size_t)Bn * NS;            // [1024, 64]
    float* sn   = dn  + (size_t)Bn * 64;            // [994, 64]
    row_norm<<<ceil_div((size_t)Bn, 128), 128, 0, stream>>>(drug, dn, Bn, 64);
    row_norm<<<ceil_div((size_t)NS, 128), 128, 0, stream>>>(se,   sn, NS, 64);
    launch_gemm(dn, sn, nullptr, freq, Bn, NS, 64, 0, 5.0f, /*BT*/true, stream);
}
